// GCNRegreesion_64467459113444
// MI455X (gfx1250) — matrix operand order repaired and checked by reading, not compiled, not run on hardware
//
#include <hip/hip_runtime.h>

typedef __attribute__((ext_vector_type(16))) _Float16     v16h;
typedef __attribute__((ext_vector_type(8)))  float        v8f;
typedef __attribute__((ext_vector_type(4)))  unsigned int u32x4;

// ---------------- graph normalization ----------------
__global__ void k_fill1(float* deg, int n) {
  int i = blockIdx.x * blockDim.x + threadIdx.x;
  if (i < n) deg[i] = 1.0f;                       // self-loop contributes 1
}

__global__ void k_deg(const long long* __restrict__ dst, float* deg, int e) {
  int i = blockIdx.x * blockDim.x + threadIdx.x;
  if (i < e) atomicAdd(&deg[(int)dst[i]], 1.0f);
}

__global__ void k_dinv(float* d, int n) {
  int i = blockIdx.x * blockDim.x + threadIdx.x;
  if (i < n) d[i] = rsqrtf(d[i]);
}

__global__ void k_enorm(const long long* __restrict__ src, const long long* __restrict__ dst,
                        const float* __restrict__ dinv, float* __restrict__ enorm, int e) {
  int i = blockIdx.x * blockDim.x + threadIdx.x;
  if (i < e) enorm[i] = dinv[(int)src[i]] * dinv[(int)dst[i]];
}

// ---------------- f32 -> f16 staging (activations) ----------------
__global__ void k_f2h(const float* __restrict__ in, _Float16* __restrict__ out, int n) {
  int i = blockIdx.x * blockDim.x + threadIdx.x;
  if (i < n) out[i] = (_Float16)in[i];
}

// ---------------- weight pre-pack: f32 KxF row-major -> WMMA B-fragment order --
// One packed uint per thread. Fragment order: for tile (k0,n0), 32 lanes x 8 VGPRs
// of packed {B[kk][col], B[kk+1][col]} with kk = k0 + 16*(lane/16) + 2v,
// col = n0 + lane%16 (contiguous k order, same as the A fragment). Fuses f32->f16.
template <int K, int F>
__global__ void k_packB(const float* __restrict__ W, unsigned int* __restrict__ wp) {
  constexpr int TILES = (K / 32) * (F / 16);
  int idx = blockIdx.x * blockDim.x + threadIdx.x;
  if (idx >= TILES * 256) return;
  const int tile = idx >> 8;
  const int r    = idx & 255;
  const int lane = r >> 3;
  const int v    = r & 7;
  const int k0   = (tile / (F / 16)) * 32;
  const int n0   = (tile % (F / 16)) * 16;
  const int kk   = k0 + 16 * (lane >> 4) + 2 * v;
  const int col  = n0 + (lane & 15);
  union { _Float16 f[2]; unsigned int u; } pk;
  pk.f[0] = (_Float16)W[(size_t)kk * F + col];
  pk.f[1] = (_Float16)W[(size_t)(kk + 1) * F + col];
  wp[idx] = pk.u;
}

// ---------------- WMMA GEMM: C[MxF] = A[MxK] * B[KxF], fp16 in / fp32 out ----
// One wave per 16-row tile; A row-major f16, B pre-packed in fragment order.
// All fragment traffic is b128 loads. M must be a multiple of 16 (50000 = 3125*16).
template <int K, int F>
__global__ void __launch_bounds__(128)
k_gemm_wmma(const _Float16* __restrict__ A, const unsigned int* __restrict__ Bp,
            float* __restrict__ C, int M) {
  const int wave = blockIdx.x * (blockDim.x >> 5) + (threadIdx.x >> 5);
  const int lane = threadIdx.x & 31;
  const int m0   = wave * 16;
  if (m0 >= M) return;                       // wave-uniform: EXEC stays all-ones
  const int half_id = lane >> 4;
  const int l16     = lane & 15;
  const unsigned int* arow = (const unsigned int*)(A + (size_t)(m0 + l16) * K);

  union Frag { v16h h; u32x4 u4[2]; };

#pragma unroll
  for (int n0 = 0; n0 < F; n0 += 16) {
    v8f acc = {};
#pragma unroll
    for (int k0 = 0; k0 < K; k0 += 32) {
      // A fragment: K = {k0+16h .. k0+16h+15}, 2x16B
      Frag a;
      a.u4[0] = *(const u32x4*)(arow + ((k0 + 16 * half_id) >> 1));
      a.u4[1] = *(const u32x4*)(arow + ((k0 + 16 * half_id + 8) >> 1));
      // B fragment: contiguous 8 uints per lane from packed buffer, 2x16B
      const unsigned int* bp =
          Bp + ((size_t)((k0 >> 5) * (F / 16) + (n0 >> 4)) << 8) + lane * 8;
      Frag b;
      b.u4[0] = *(const u32x4*)(bp);
      b.u4[1] = *(const u32x4*)(bp + 4);
      acc = __builtin_amdgcn_wmma_f32_16x16x32_f16(false, a.h, false, b.h,
                                                   (short)0, acc, false, false);
    }
    // C/D layout: VGPR r holds row m0 + 8*half_id + r, column n0 + l16
    const int col = n0 + l16;
#pragma unroll
    for (int r = 0; r < 8; ++r)
      C[(size_t)(m0 + 8 * half_id + r) * F + col] = acc[r];
  }
}

// ---------------- aggregation ----------------
// self-loop: agg[i,f] = t[i,f] * dinv[i]^2
__global__ void k_selfagg(const float* __restrict__ t, const float* __restrict__ dinv,
                          float* __restrict__ agg, int total, int F) {
  int i = blockIdx.x * blockDim.x + threadIdx.x;
  if (i >= total) return;
  const float dv = dinv[i / F];
  agg[i] = t[i] * dv * dv;
}

// per-edge scatter: agg[dst] += t[src] * enorm[e]; one thread = 1 edge x 4 features
__global__ void k_scatter(const long long* __restrict__ src, const long long* __restrict__ dst,
                          const float* __restrict__ enorm, const float* __restrict__ t,
                          float* __restrict__ agg, int E, int F4) {
  int idx = blockIdx.x * blockDim.x + threadIdx.x;
  if (idx >= E * F4) return;
  const int e = idx / F4;
  const int q = idx - e * F4;
  const int s = (int)src[e];
  const int d = (int)dst[e];
  const float w = enorm[e];
  const float4 v = *(const float4*)(t + (size_t)s * (F4 * 4) + q * 4);
  float* ap = agg + (size_t)d * (F4 * 4) + q * 4;
  atomicAdd(ap + 0, v.x * w);
  atomicAdd(ap + 1, v.y * w);
  atomicAdd(ap + 2, v.z * w);
  atomicAdd(ap + 3, v.w * w);
}

// bias + relu, emit fp16 for the next WMMA GEMM
__global__ void k_bias_relu_h(const float* __restrict__ agg, const float* __restrict__ bias,
                              _Float16* __restrict__ outh, int total, int fmask) {
  int i = blockIdx.x * blockDim.x + threadIdx.x;
  if (i >= total) return;
  float v = agg[i] + bias[i & fmask];
  outh[i] = (_Float16)(v > 0.0f ? v : 0.0f);
}

// fused layer-3 bias/relu + 32->1 linear head
__global__ void k_final(const float* __restrict__ agg, const float* __restrict__ b3,
                        const float* __restrict__ Wl, const float* __restrict__ bl,
                        float* __restrict__ out, int n) {
  int i = blockIdx.x * blockDim.x + threadIdx.x;
  if (i >= n) return;
  const float* row = agg + (size_t)i * 32;
  float s = bl[0];
#pragma unroll
  for (int f = 0; f < 32; ++f) {
    float v = row[f] + b3[f];
    s += (v > 0.0f ? v : 0.0f) * Wl[f];
  }
  out[i] = s;
}

static inline int cdiv(long long a, long long b) { return (int)((a + b - 1) / b); }

extern "C" void kernel_launch(void* const* d_in, const int* in_sizes, int n_in,
                              void* d_out, int out_size, void* d_ws, size_t ws_size,
                              hipStream_t stream) {
  const float*     x   = (const float*)d_in[0];
  const long long* ei  = (const long long*)d_in[1];
  const float*     W1  = (const float*)d_in[2];
  const float*     b1  = (const float*)d_in[3];
  const float*     W2  = (const float*)d_in[4];
  const float*     b2  = (const float*)d_in[5];
  const float*     W3  = (const float*)d_in[6];
  const float*     b3  = (const float*)d_in[7];
  const float*     Wl  = (const float*)d_in[8];
  const float*     bl  = (const float*)d_in[9];
  float*           out = (float*)d_out;

  const int N = in_sizes[0] / 128;   // 50000 (multiple of 16)
  const int E = in_sizes[1] / 2;     // 800000
  const long long* srcI = ei;
  const long long* dstI = ei + E;

  // workspace carve-out (256B aligned)
  char* p = (char*)d_ws;
  size_t off = 0;
  auto take = [&](size_t bytes) -> char* {
    char* r = p + off;
    off += (bytes + 255) & ~(size_t)255;
    return r;
  };
  float*        dinv  = (float*)take((size_t)N * 4);
  float*        enorm = (float*)take((size_t)E * 4);
  float*        tbuf  = (float*)take((size_t)N * 128 * 4);
  float*        agg   = (float*)take((size_t)N * 128 * 4);
  _Float16*     ahalf = (_Float16*)take((size_t)N * 128 * 2);
  unsigned int* wpack = (unsigned int*)take((size_t)8192 * 4);  // max (128/32)*(128/16)*256
  (void)ws_size; (void)n_in; (void)out_size;

  const int T = 256;
  const int gemm_blocks = cdiv(cdiv(N, 16), 4);  // 4 waves per 128-thread block

  // ---- normalization (shared across layers) ----
  k_fill1<<<cdiv(N, T), T, 0, stream>>>(dinv, N);
  k_deg  <<<cdiv(E, T), T, 0, stream>>>(dstI, dinv, E);
  k_dinv <<<cdiv(N, T), T, 0, stream>>>(dinv, N);
  k_enorm<<<cdiv(E, T), T, 0, stream>>>(srcI, dstI, dinv, enorm, E);

  // ---- layer 1: 128 -> 128 ----
  k_f2h<<<cdiv((long long)N * 128, T), T, 0, stream>>>(x, ahalf, N * 128);
  k_packB<128, 128><<<cdiv(8192, T), T, 0, stream>>>(W1, wpack);
  k_gemm_wmma<128, 128><<<gemm_blocks, 128, 0, stream>>>(ahalf, wpack, tbuf, N);
  k_selfagg<<<cdiv((long long)N * 128, T), T, 0, stream>>>(tbuf, dinv, agg, N * 128, 128);
  k_scatter<<<cdiv((long long)E * 32, T), T, 0, stream>>>(srcI, dstI, enorm, tbuf, agg, E, 32);
  k_bias_relu_h<<<cdiv((long long)N * 128, T), T, 0, stream>>>(agg, b1, ahalf, N * 128, 127);

  // ---- layer 2: 128 -> 64 ----
  k_packB<128, 64><<<cdiv(4096, T), T, 0, stream>>>(W2, wpack);
  k_gemm_wmma<128, 64><<<gemm_blocks, 128, 0, stream>>>(ahalf, wpack, tbuf, N);
  k_selfagg<<<cdiv((long long)N * 64, T), T, 0, stream>>>(tbuf, dinv, agg, N * 64, 64);
  k_scatter<<<cdiv((long long)E * 16, T), T, 0, stream>>>(srcI, dstI, enorm, tbuf, agg, E, 16);
  k_bias_relu_h<<<cdiv((long long)N * 64, T), T, 0, stream>>>(agg, b2, ahalf, N * 64, 63);

  // ---- layer 3: 64 -> 32 ----
  k_packB<64, 32><<<cdiv(1024, T), T, 0, stream>>>(W3, wpack);
  k_gemm_wmma<64, 32><<<gemm_blocks, 128, 0, stream>>>(ahalf, wpack, tbuf, N);
  k_selfagg<<<cdiv((long long)N * 32, T), T, 0, stream>>>(tbuf, dinv, agg, N * 32, 32);
  k_scatter<<<cdiv((long long)E * 8, T), T, 0, stream>>>(srcI, dstI, enorm, tbuf, agg, E, 8);

  // ---- fused bias3 + relu + linear head (32 -> 1) ----
  k_final<<<cdiv(N, T), T, 0, stream>>>(agg, b3, Wl, bl, out, N);
}
